// AttentionBlock_57320633532572
// MI455X (gfx1250) — compile-verified, matrix operand order checked by reading, not run on hardware
//
#include <hip/hip_runtime.h>
#include <hip/hip_bf16.h>

// ---------------------------------------------------------------------------
// Attention block for MI455X (gfx1250): bf16 WMMA GEMMs + column softmax.
//   q = x@Wq+bq ; k = x@Wk+bk ; v = x@Wv+bv
//   scores = q@k^T * 1/sqrt(1024) ; attn = softmax(scores, axis=QUERY)
//   out = attn @ v
// All GEMMs: C[M,N] = A[M,K] (row-major bf16) x Bt[N,K] (row-major bf16).
// Double-buffered LDS; async global->LDS copies (ASYNCcnt) when available.
// ---------------------------------------------------------------------------

typedef __attribute__((ext_vector_type(16))) __bf16 v16bf;
typedef __attribute__((ext_vector_type(8)))  __bf16 v8bf;
typedef __attribute__((ext_vector_type(4)))  __bf16 v4bf;
typedef __attribute__((ext_vector_type(8)))  float  v8f;
typedef __attribute__((ext_vector_type(4)))  int    v4i;

#define BATCH 8
#define SEQ   2048
#define EMB   1024
#define MTOT  (BATCH * SEQ)   // 16384

constexpr int BM = 128, BN = 128, BK = 32, LDST = BK + 8; // pad: conflict-free b128
constexpr int TBUF = BM * LDST;                           // elems per LDS tile buffer

#if defined(__has_builtin)
#if __has_builtin(__builtin_amdgcn_global_load_async_to_lds_b128)
#define USE_ASYNC_LDS 1
#endif
#endif

// 16-byte global -> LDS copy (async DMA when the gfx1250 builtin exists).
// Builtin signature (from hipcc diagnostic): (v4i AS1*, v4i AS3*, imm, imm).
__device__ __forceinline__ void copy16_g2l(__bf16* dst, const __bf16* src) {
#ifdef USE_ASYNC_LDS
  __builtin_amdgcn_global_load_async_to_lds_b128(
      (__attribute__((address_space(1))) v4i*)(src),
      (__attribute__((address_space(3))) v4i*)(dst), 0, 0);
#else
  *reinterpret_cast<v8bf*>(dst) = *reinterpret_cast<const v8bf*>(src);
#endif
}

__device__ __forceinline__ void wait_async_copies() {
#ifdef USE_ASYNC_LDS
#if __has_builtin(__builtin_amdgcn_s_wait_asynccnt)
  __builtin_amdgcn_s_wait_asynccnt(0);
#else
  asm volatile("s_wait_asynccnt 0x0" ::: "memory");
#endif
#endif
}

// Stage one 128x32 A tile and one 128x32 Bt tile into LDS buffers.
__device__ __forceinline__ void stage_tiles(const __bf16* __restrict__ A,
                                            const __bf16* __restrict__ Bt,
                                            __bf16* as, __bf16* bs,
                                            int rowBase, int colBase,
                                            int lda, int ldb, int k0,
                                            int ldRow, int ldCol) {
#pragma unroll
  for (int it = 0; it < 2; ++it) {
    int r = ldRow + it * 64;
    copy16_g2l(&as[r * LDST + ldCol],
               &A[(long long)(rowBase + r) * lda + k0 + ldCol]);
    copy16_g2l(&bs[r * LDST + ldCol],
               &Bt[(long long)(colBase + r) * ldb + k0 + ldCol]);
  }
}

// -------------------------- fp32 -> bf16 cast ------------------------------
__global__ void cast_f32_bf16_kernel(const float* __restrict__ in,
                                     __bf16* __restrict__ out, int n4) {
  int i = blockIdx.x * blockDim.x + threadIdx.x;
  if (i < n4) {
    float4 f = reinterpret_cast<const float4*>(in)[i];
    v4bf o;
    o[0] = (__bf16)f.x; o[1] = (__bf16)f.y; o[2] = (__bf16)f.z; o[3] = (__bf16)f.w;
    reinterpret_cast<v4bf*>(out)[i] = o;
  }
}

// -------------------- W[K,N] fp32 -> W^T[N,K] bf16 -------------------------
__global__ void transpose_cast_kernel(const float* __restrict__ W,
                                      __bf16* __restrict__ WT,
                                      int Kdim, int Ndim) {
  __shared__ __bf16 t[32][33];
  int n0 = blockIdx.x * 32, k0 = blockIdx.y * 32;
  for (int j = threadIdx.y; j < 32; j += 8)
    t[j][threadIdx.x] = (__bf16)W[(long long)(k0 + j) * Ndim + n0 + threadIdx.x];
  __syncthreads();
  for (int j = threadIdx.y; j < 32; j += 8)
    WT[(long long)(n0 + j) * Kdim + k0 + threadIdx.x] = t[threadIdx.x][j];
}

// --------------------------- bf16 WMMA GEMM --------------------------------
// BIAS_MODE: 0 = none, 1 = bias[n] (per output column), 2 = bias[m] (per row)
template <int BIAS_MODE, bool OUT_BF16>
__global__ __launch_bounds__(256)
void gemm_bf16_wmma(const __bf16* __restrict__ A, const __bf16* __restrict__ Bt,
                    const float* __restrict__ bias, void* __restrict__ Cout,
                    int K, int lda, int ldb, int ldc,
                    long long bsA, long long bsB, long long bsC, float scale) {
  __shared__ __align__(16) __bf16 As[2 * TBUF];
  __shared__ __align__(16) __bf16 Bs[2 * TBUF];

  A  += (long long)blockIdx.z * bsA;
  Bt += (long long)blockIdx.z * bsB;

  const int tid  = threadIdx.x;
  const int lane = tid & 31;
  const int wave = tid >> 5;          // 8 waves
  const int wm   = wave >> 2;         // 0..1  (64 rows each)
  const int wn   = wave & 3;          // 0..3  (32 cols each)
  const int lh   = lane >> 4;         // lane half (ISA fragment layout)
  const int lr   = lane & 15;

  const int rowBase = blockIdx.y * BM;
  const int colBase = blockIdx.x * BN;

  v8f acc[4][2] = {};

  const int ldRow = tid >> 2;         // 0..63
  const int ldCol = (tid & 3) * 8;    // 0,8,16,24

  // Prologue: fill buffer 0 with the first K-chunk.
  stage_tiles(A, Bt, &As[0], &Bs[0], rowBase, colBase, lda, ldb, 0, ldRow, ldCol);
  wait_async_copies();
  __syncthreads();

  int cur = 0;
  for (int k0 = 0; k0 < K; k0 += BK) {
    // Kick off the DMA for the next chunk into the other buffer while we
    // compute on this one (overlaps ASYNCcnt traffic with WMMA issue).
    if (k0 + BK < K)
      stage_tiles(A, Bt, &As[(cur ^ 1) * TBUF], &Bs[(cur ^ 1) * TBUF],
                  rowBase, colBase, lda, ldb, k0 + BK, ldRow, ldCol);

    const __bf16* as = &As[cur * TBUF];
    const __bf16* bs = &Bs[cur * TBUF];

    // ---- build fragments per ISA 16-bit A/B 16x32 layout ----
    // lane (lr + 16*lh): elems[0..7] = K(8*lh .. +8), elems[8..15] = K(16+8*lh .. +8)
    v16bf af[4], bfr[2];
#pragma unroll
    for (int mt = 0; mt < 4; ++mt) {
      const __bf16* p = &as[(wm * 64 + mt * 16 + lr) * LDST + lh * 8];
      v8bf lo = *reinterpret_cast<const v8bf*>(p);
      v8bf hi = *reinterpret_cast<const v8bf*>(p + 16);
      af[mt] = __builtin_shufflevector(lo, hi, 0,1,2,3,4,5,6,7,8,9,10,11,12,13,14,15);
    }
#pragma unroll
    for (int nt = 0; nt < 2; ++nt) {
      const __bf16* p = &bs[(wn * 32 + nt * 16 + lr) * LDST + lh * 8];
      v8bf lo = *reinterpret_cast<const v8bf*>(p);
      v8bf hi = *reinterpret_cast<const v8bf*>(p + 16);
      bfr[nt] = __builtin_shufflevector(lo, hi, 0,1,2,3,4,5,6,7,8,9,10,11,12,13,14,15);
    }

#pragma unroll
    for (int mt = 0; mt < 4; ++mt)
#pragma unroll
      for (int nt = 0; nt < 2; ++nt)
        acc[mt][nt] = __builtin_amdgcn_wmma_f32_16x16x32_bf16(
            false, af[mt], false, bfr[nt], (short)0, acc[mt][nt], false, false);

    // Our async writes into buffer cur^1 must land, and every wave must be
    // done reading buffer cur, before the roles swap. One barrier per chunk.
    wait_async_copies();
    __syncthreads();
    cur ^= 1;
  }

  // ---- epilogue: C/D 16x16 f32 layout: lane -> N=lr, VGPR r -> M=8*lh+r ----
  long long cbase = (long long)blockIdx.z * bsC;
#pragma unroll
  for (int mt = 0; mt < 4; ++mt)
#pragma unroll
    for (int nt = 0; nt < 2; ++nt) {
      int gn = colBase + wn * 32 + nt * 16 + lr;
#pragma unroll
      for (int r = 0; r < 8; ++r) {
        int gm = rowBase + wm * 64 + mt * 16 + lh * 8 + r;
        float v = acc[mt][nt][r] * scale;
        if (BIAS_MODE == 1) v += bias[gn];
        if (BIAS_MODE == 2) v += bias[gm];
        long long idx = cbase + (long long)gm * ldc + gn;
        if constexpr (OUT_BF16)
          reinterpret_cast<__bf16*>(Cout)[idx] = (__bf16)v;
        else
          reinterpret_cast<float*>(Cout)[idx] = v;
      }
    }
}

// ------------- softmax over the QUERY axis (column softmax) ----------------
// scores[b][q][k] fp32 -> attn[b][q][k] bf16, normalized over q for each (b,k).
// Thread owns one column k; all passes coalesced across lanes.
__global__ void softmax_col_kernel(const float* __restrict__ scores,
                                   __bf16* __restrict__ attn, int S) {
  int col = blockIdx.x * blockDim.x + threadIdx.x;
  const float* base = scores + (long long)blockIdx.y * S * S + col;
  __bf16* ob = attn + (long long)blockIdx.y * S * S + col;

  float m = -3.0e38f;
  for (int q = 0; q < S; ++q) m = fmaxf(m, base[(long long)q * S]);
  float s = 0.f;
  for (int q = 0; q < S; ++q) s += __expf(base[(long long)q * S] - m);
  float inv = 1.f / s;
  for (int q = 0; q < S; ++q)
    ob[(long long)q * S] = (__bf16)(__expf(base[(long long)q * S] - m) * inv);
}

// ---------------------------------------------------------------------------
extern "C" void kernel_launch(void* const* d_in, const int* in_sizes, int n_in,
                              void* d_out, int out_size, void* d_ws, size_t ws_size,
                              hipStream_t stream) {
  const float* x  = (const float*)d_in[0];
  const float* Wq = (const float*)d_in[1];
  const float* bq = (const float*)d_in[2];
  const float* Wk = (const float*)d_in[3];
  const float* bk = (const float*)d_in[4];
  const float* Wv = (const float*)d_in[5];
  const float* bv = (const float*)d_in[6];
  float* out = (float*)d_out;

  // -------- workspace layout (all chunks 256B-aligned by construction) -----
  char* w = (char*)d_ws;
  size_t o = 0;
  __bf16* xbf = (__bf16*)(w + o); o += (size_t)MTOT * EMB * 2;   // 32 MB
  __bf16* Qbf = (__bf16*)(w + o); o += (size_t)MTOT * EMB * 2;   // 32 MB
  __bf16* Kbf = (__bf16*)(w + o); o += (size_t)MTOT * EMB * 2;   // 32 MB
  __bf16* Vt  = (__bf16*)(w + o); o += (size_t)EMB * MTOT * 2;   // 32 MB (V^T [E, B*S])
  __bf16* WqT = (__bf16*)(w + o); o += (size_t)EMB * EMB * 2;    // 2 MB
  __bf16* WkT = (__bf16*)(w + o); o += (size_t)EMB * EMB * 2;
  __bf16* WvT = (__bf16*)(w + o); o += (size_t)EMB * EMB * 2;
  float*  scoresF = (float*)(w + o);  o += (size_t)BATCH * SEQ * SEQ * 4; // 128 MB
  __bf16* attnbf  = (__bf16*)(w + o); o += (size_t)BATCH * SEQ * SEQ * 2; //  64 MB
  (void)ws_size; (void)o; (void)in_sizes; (void)n_in; (void)out_size;

  dim3 blk(256);

  // 1) cast x -> bf16
  int n4 = MTOT * EMB / 4;
  cast_f32_bf16_kernel<<<dim3(n4 / 256), blk, 0, stream>>>(x, xbf, n4);

  // 2) transpose+cast weights -> W^T bf16 [N,K]
  dim3 tb(32, 8), tg(EMB / 32, EMB / 32);
  transpose_cast_kernel<<<tg, tb, 0, stream>>>(Wq, WqT, EMB, EMB);
  transpose_cast_kernel<<<tg, tb, 0, stream>>>(Wk, WkT, EMB, EMB);
  transpose_cast_kernel<<<tg, tb, 0, stream>>>(Wv, WvT, EMB, EMB);

  // 3) Q = x@Wq+bq, K = x@Wk+bk  (bf16 out, row-major [B*S, E])
  gemm_bf16_wmma<1, true><<<dim3(EMB / 128, MTOT / 128, 1), blk, 0, stream>>>(
      xbf, WqT, bq, Qbf, EMB, EMB, EMB, EMB, 0, 0, 0, 1.0f);
  gemm_bf16_wmma<1, true><<<dim3(EMB / 128, MTOT / 128, 1), blk, 0, stream>>>(
      xbf, WkT, bk, Kbf, EMB, EMB, EMB, EMB, 0, 0, 0, 1.0f);

  // 4) V^T = Wv^T @ x^T + bv (row bias): A=WvT[E,E], Bt=xbf[B*S,E], C=Vt[E,B*S]
  gemm_bf16_wmma<2, true><<<dim3(MTOT / 128, EMB / 128, 1), blk, 0, stream>>>(
      WvT, xbf, bv, Vt, EMB, EMB, EMB, MTOT, 0, 0, 0, 1.0f);

  // 5) scores_b = Q_b @ K_b^T * 1/32   (Bt = K row-major directly)
  gemm_bf16_wmma<0, false><<<dim3(SEQ / 128, SEQ / 128, BATCH), blk, 0, stream>>>(
      Qbf, Kbf, nullptr, scoresF, EMB, EMB, EMB, SEQ,
      (long long)SEQ * EMB, (long long)SEQ * EMB, (long long)SEQ * SEQ, 0.03125f);

  // 6) attn = softmax over query axis (per-column), emit bf16
  softmax_col_kernel<<<dim3(SEQ / 256, BATCH), blk, 0, stream>>>(scoresF, attnbf, SEQ);

  // 7) out_b = attn_b @ V_b : A=attn[S,S], Bt = Vt slice [E, S] (ldb = B*S)
  gemm_bf16_wmma<0, false><<<dim3(EMB / 128, SEQ / 128, BATCH), blk, 0, stream>>>(
      attnbf, Vt, nullptr, out, SEQ, SEQ, MTOT, EMB,
      (long long)SEQ * SEQ, (long long)SEQ, (long long)SEQ * EMB, 1.0f);
}
